// THGAT_17901423690202
// MI455X (gfx1250) — matrix operand order repaired and checked by reading, not compiled, not run on hardware
//
#include <hip/hip_runtime.h>

typedef __attribute__((ext_vector_type(16))) __bf16 v16bf;
typedef __attribute__((ext_vector_type(8)))  float  v8f;

#define B_    4
#define N_    4096
#define E_    2048
#define CX_   64
#define CM_   32
#define ALPHA_ 0.2f
#define NEG_  -9.0e15f

__device__ __forceinline__ float leaky(float x) { return x > 0.f ? x : ALPHA_ * x; }

// ---------------------------------------------------------------------------
// s[b,n] = sum_c cat(x,m,pri_n)[b,c,n] * weight[c] + bias[n]
// ---------------------------------------------------------------------------
__global__ void k_node_scalar(const float* __restrict__ x, const float* __restrict__ m,
                              const float* __restrict__ pn, const float* __restrict__ w,
                              const float* __restrict__ bias, float* __restrict__ s) {
    int idx = blockIdx.x * blockDim.x + threadIdx.x;
    if (idx >= B_ * N_) return;
    int b = idx / N_, n = idx % N_;
    float acc = bias[n];
    const float* xb = x  + (size_t)b * CX_ * N_ + n;
    const float* mb = m  + (size_t)b * CM_ * N_ + n;
    const float* pb = pn + (size_t)b * CM_ * N_ + n;   // CN == CM == 32
#pragma unroll 4
    for (int c = 0; c < CX_; ++c) acc += xb[(size_t)c * N_] * w[c];
#pragma unroll 4
    for (int c = 0; c < CM_; ++c) acc += mb[(size_t)c * N_] * w[CX_ + c];
#pragma unroll 4
    for (int c = 0; c < CM_; ++c) acc += pb[(size_t)c * N_] * w[CX_ + CM_ + c];
    s[idx] = acc;
}

// ---------------------------------------------------------------------------
// deg[e] = sum_n incidence[n,e]   (column sums; coalesced across e)
// ---------------------------------------------------------------------------
__global__ void k_deg(const float* __restrict__ inc, float* __restrict__ deg) {
    int e = blockIdx.x * blockDim.x + threadIdx.x;
    if (e >= E_) return;
    float d = 0.f;
    for (int n = 0; n < N_; ++n) d += inc[(size_t)n * E_ + e];
    deg[e] = d;
}

// ---------------------------------------------------------------------------
// GEMM1: edge0[b,e] = (sum_k s[b,k] * inc[k,e]) / deg[e]
// One wave per 16-column tile. M padded 4 -> 16 with zero rows.
// bf16 WMMA, fp32 accumulate: v_wmma_f32_16x16x32_bf16.
// A layout (16-bit A 16x32): lane half h, row M=lane&15; V0..V3 = K[8h..8h+7],
// V4..V7 = K[16+8h..16+8h+7]. B layout: lane half h, col N=lane&15, same K
// order as A (two packed per VGPR).
// ---------------------------------------------------------------------------
__global__ void k_gemm_edge0(const float* __restrict__ s, const float* __restrict__ inc,
                             const float* __restrict__ deg, float* __restrict__ edge0) {
    const int lane = threadIdx.x;       // blockDim.x == 32, one wave
    const int h   = lane >> 4;
    const int col = lane & 15;
    const int e   = blockIdx.x * 16 + col;
    v8f acc = {};
    for (int k0 = 0; k0 < N_; k0 += 32) {
        v16bf a = {};                   // zero rows M >= 4
        if (col < B_) {
            const float* ar = s + (size_t)col * N_ + k0;
#pragma unroll
            for (int i = 0; i < 8; ++i) a[i]     = (__bf16)ar[8 * h + i];
#pragma unroll
            for (int i = 0; i < 8; ++i) a[8 + i] = (__bf16)ar[16 + 8 * h + i];
        }
        v16bf bf;
#pragma unroll
        for (int i = 0; i < 16; ++i)
            bf[i] = (__bf16)inc[(size_t)(k0 + 8 * h + i + (i & 8)) * E_ + e];
        acc = __builtin_amdgcn_wmma_f32_16x16x32_bf16(false, a, false, bf,
                                                      (short)0, acc, false, false);
    }
    float invd = 1.0f / deg[e];
#pragma unroll
    for (int r = 0; r < 8; ++r) {
        int row = 8 * h + r;            // C/D layout: VGPR r -> M = 8h + r
        if (row < B_) edge0[(size_t)row * E_ + e] = acc[r] * invd;
    }
}

// ---------------------------------------------------------------------------
// GEMM2: edge[b,e] = sum_k edge0[b,k] * weight2[k,e]
// Also emits edge to d_out and time_e = pri_e[b,e,1] + edge.
// ---------------------------------------------------------------------------
__global__ void k_gemm_edge(const float* __restrict__ edge0, const float* __restrict__ w2,
                            const float* __restrict__ pe, float* __restrict__ edge,
                            float* __restrict__ t, float* __restrict__ out_edge) {
    const int lane = threadIdx.x;
    const int h   = lane >> 4;
    const int col = lane & 15;
    const int e   = blockIdx.x * 16 + col;
    v8f acc = {};
    for (int k0 = 0; k0 < E_; k0 += 32) {
        v16bf a = {};
        if (col < B_) {
            const float* ar = edge0 + (size_t)col * E_ + k0;
#pragma unroll
            for (int i = 0; i < 8; ++i) a[i]     = (__bf16)ar[8 * h + i];
#pragma unroll
            for (int i = 0; i < 8; ++i) a[8 + i] = (__bf16)ar[16 + 8 * h + i];
        }
        v16bf bf;
#pragma unroll
        for (int i = 0; i < 16; ++i)
            bf[i] = (__bf16)w2[(size_t)(k0 + 8 * h + i + (i & 8)) * E_ + e];
        acc = __builtin_amdgcn_wmma_f32_16x16x32_bf16(false, a, false, bf,
                                                      (short)0, acc, false, false);
    }
#pragma unroll
    for (int r = 0; r < 8; ++r) {
        int row = 8 * h + r;
        if (row < B_) {
            float v = acc[r];
            size_t o = (size_t)row * E_ + e;
            edge[o]     = v;
            out_edge[o] = v;                       // edge output [B,E,1]
            t[o]        = pe[o * 2 + 1] + v;       // pri_e[:,:,1] + edge
        }
    }
}

// ---------------------------------------------------------------------------
// Softmax stats over nodes per (b,e): m = max, den = sum exp(score - m).
// score(n) = leaky(a0*s[b,n] + a1*edge[b,e]) + t[b,e] if inc[n,e] > 0.
// Writes u = t - m, w = edge / den  (per (b,e)).
// ---------------------------------------------------------------------------
__global__ void k_stats(const float* __restrict__ inc, const float* __restrict__ s,
                        const float* __restrict__ edge, const float* __restrict__ t,
                        const float* __restrict__ a2, float* __restrict__ u,
                        float* __restrict__ wq) {
    int e = blockIdx.x * blockDim.x + threadIdx.x;
    if (e >= E_) return;
    const float a0 = a2[0], a1 = a2[1];
    float cb[B_], tb[B_], mb[B_], den[B_];
#pragma unroll
    for (int b = 0; b < B_; ++b) {
        cb[b] = edge[(size_t)b * E_ + e];
        tb[b] = t[(size_t)b * E_ + e];
        mb[b] = NEG_;
        den[b] = 0.f;
    }
    for (int n = 0; n < N_; ++n) {                 // pass 1: max
        float iv = inc[(size_t)n * E_ + e];
        if (iv > 0.f) {
#pragma unroll
            for (int b = 0; b < B_; ++b) {
                float sc = leaky(a0 * s[(size_t)b * N_ + n] + a1 * cb[b]) + tb[b];
                mb[b] = fmaxf(mb[b], sc);
            }
        }
    }
    for (int n = 0; n < N_; ++n) {                 // pass 2: sum of exp (L2-hot)
        float iv = inc[(size_t)n * E_ + e];
        if (iv > 0.f) {
#pragma unroll
            for (int b = 0; b < B_; ++b) {
                float sc = leaky(a0 * s[(size_t)b * N_ + n] + a1 * cb[b]) + tb[b];
                den[b] += __expf(sc - mb[b]);
            }
        }
    }
#pragma unroll
    for (int b = 0; b < B_; ++b) {
        u[(size_t)b * E_ + e]  = tb[b] - mb[b];
        wq[(size_t)b * E_ + e] = cb[b] / den[b];
    }
}

// ---------------------------------------------------------------------------
// node[b,n] = sum_e [inc>0] exp(leaky(a0*s + a1*edge) + u[b,e]) * w[b,e]
// One block per n; 256 threads stride over e (coalesced incidence row reads);
// LDS tree reduction for the 4 batch accumulators.
// ---------------------------------------------------------------------------
__global__ void k_node(const float* __restrict__ inc, const float* __restrict__ s,
                       const float* __restrict__ edge, const float* __restrict__ u,
                       const float* __restrict__ wq, const float* __restrict__ a2,
                       float* __restrict__ out_node) {
    const int n = blockIdx.x;
    const int tid = threadIdx.x;
    const float a0 = a2[0], a1 = a2[1];
    float sb[B_];
#pragma unroll
    for (int b = 0; b < B_; ++b) sb[b] = s[(size_t)b * N_ + n];
    float part[B_] = {0.f, 0.f, 0.f, 0.f};
    for (int e = tid; e < E_; e += 256) {
        float iv = inc[(size_t)n * E_ + e];
        if (iv > 0.f) {
#pragma unroll
            for (int b = 0; b < B_; ++b) {
                size_t o = (size_t)b * E_ + e;
                float c = edge[o];
                float arg = leaky(a0 * sb[b] + a1 * c) + u[o];
                part[b] += __expf(arg) * wq[o];
            }
        }
    }
    __shared__ float red[B_ * 256];
#pragma unroll
    for (int b = 0; b < B_; ++b) red[b * 256 + tid] = part[b];
    __syncthreads();
    for (int off = 128; off > 0; off >>= 1) {
        if (tid < off) {
#pragma unroll
            for (int b = 0; b < B_; ++b)
                red[b * 256 + tid] += red[b * 256 + tid + off];
        }
        __syncthreads();
    }
    if (tid == 0) {
#pragma unroll
        for (int b = 0; b < B_; ++b) out_node[(size_t)b * N_ + n] = red[b * 256];
    }
}

// ---------------------------------------------------------------------------
extern "C" void kernel_launch(void* const* d_in, const int* in_sizes, int n_in,
                              void* d_out, int out_size, void* d_ws, size_t ws_size,
                              hipStream_t stream) {
    (void)in_sizes; (void)n_in; (void)out_size; (void)ws_size;
    const float* x    = (const float*)d_in[0];
    const float* m    = (const float*)d_in[1];
    const float* pn   = (const float*)d_in[2];
    const float* pe   = (const float*)d_in[3];
    const float* inc  = (const float*)d_in[4];
    const float* wt   = (const float*)d_in[5];
    const float* w2   = (const float*)d_in[6];
    const float* a2   = (const float*)d_in[7];
    const float* bias = (const float*)d_in[8];

    float* out       = (float*)d_out;
    float* out_node  = out;              // [B,1,N] flat = b*N + n
    float* out_edge  = out + B_ * N_;    // [B,E,1] flat = b*E + e

    float* ws    = (float*)d_ws;
    float* s     = ws;  ws += B_ * N_;
    float* deg   = ws;  ws += E_;
    float* edge0 = ws;  ws += B_ * E_;
    float* edge  = ws;  ws += B_ * E_;
    float* t     = ws;  ws += B_ * E_;
    float* u     = ws;  ws += B_ * E_;
    float* wq    = ws;  ws += B_ * E_;

    k_node_scalar<<<(B_ * N_ + 255) / 256, 256, 0, stream>>>(x, m, pn, wt, bias, s);
    k_deg<<<(E_ + 255) / 256, 256, 0, stream>>>(inc, deg);
    k_gemm_edge0<<<E_ / 16, 32, 0, stream>>>(s, inc, deg, edge0);
    k_gemm_edge<<<E_ / 16, 32, 0, stream>>>(edge0, w2, pe, edge, t, out_edge);
    k_stats<<<E_ / 256, 256, 0, stream>>>(inc, s, edge, t, a2, u, wq);
    k_node<<<N_, 256, 0, stream>>>(inc, s, edge, u, wq, a2, out_node);
}
